// ResidualBlock_45054206935389
// MI455X (gfx1250) — matrix operand order repaired and checked by reading, not compiled, not run on hardware
//
#include <hip/hip_runtime.h>

// ResidualBlock + self-attention for MI455X (gfx1250), wave32 + WMMA f16 (16x16x32, fp32 acc).
// conv3d = implicit GEMM over a halo-padded channel-last volume: every tap is a
// constant-immediate-offset global_load_b128 feeding v_wmma (no exec masking, no div math).
// Attention is flash-style; the V panel is staged into LDS with the Tensor Data Mover
// (tensor_load_to_lds + s_wait_tensorcnt), double-buffered so DMA overlaps WMMA.

typedef __attribute__((ext_vector_type(16))) _Float16 v16h;
typedef __attribute__((ext_vector_type(8)))  _Float16 v8h;
typedef __attribute__((ext_vector_type(8)))  float    v8f;
typedef __attribute__((ext_vector_type(4)))  unsigned int u32x4;
typedef __attribute__((ext_vector_type(8)))  int          i32x8;
typedef __attribute__((ext_vector_type(4)))  int          i32x4;

#define DEVINL __device__ __forceinline__

#if defined(__HIP_DEVICE_COMPILE__) && __has_builtin(__builtin_amdgcn_tensor_load_to_lds) && \
    __has_builtin(__builtin_amdgcn_s_wait_tensorcnt)
#define HAVE_TDM 1
#else
#define HAVE_TDM 0
#endif

static constexpr int Bc = 2;
static constexpr int C  = 256;
static constexpr int C8 = 32;
static constexpr int Nv = 4096;   // 16*16*16 voxels
static constexpr int P1 = 18;     // padded edge
static constexpr int PV = P1 * P1 * P1;  // 5832 padded voxels

DEVINL v16h cat16(v8h lo, v8h hi) {
  return __builtin_shufflevector(lo, hi, 0,1,2,3,4,5,6,7,8,9,10,11,12,13,14,15);
}

// WMMA A operand (16x32 f16) from row-major matrix; p -> &mat[m0][k0], ld = row stride.
// lane<16: M=lane, K=0..7,16..23 ; lane>=16: M=lane-16, K=8..15,24..31. (ISA 7.12.2)
DEVINL v16h load_A_rm(const _Float16* p, int ld) {
  int lane = threadIdx.x & 31;
  int m  = lane & 15;
  int kb = (lane < 16) ? 0 : 8;
  const _Float16* q = p + (size_t)m * ld + kb;
  return cat16(*(const v8h*)(q), *(const v8h*)(q + 16));
}

// WMMA B operand (32x16 f16), K-contiguous storage: element (k,n) at p[n*ld + k].
// lane<16: n=lane, K=0..7,16..23 ; lane>=16: n=lane-16, K=8..15,24..31.
DEVINL v16h load_B_kc(const _Float16* p, int ld) {
  int lane = threadIdx.x & 31;
  int n  = lane & 15;
  int kb = (lane < 16) ? 0 : 8;
  const _Float16* q = p + (size_t)n * ld + kb;
  return cat16(*(const v8h*)(q), *(const v8h*)(q + 16));
}

DEVINL v8f wmma_f16(v16h a, v16h b, v8f c) {
  return __builtin_amdgcn_wmma_f32_16x16x32_f16(false, a, false, b, (short)0, c, false, false);
}

DEVINL float half_max16(float v) {
  v = fmaxf(v, __shfl_xor(v, 1, 32));
  v = fmaxf(v, __shfl_xor(v, 2, 32));
  v = fmaxf(v, __shfl_xor(v, 4, 32));
  v = fmaxf(v, __shfl_xor(v, 8, 32));
  return v;
}
DEVINL float half_sum16(float v) {
  v += __shfl_xor(v, 1, 32);
  v += __shfl_xor(v, 2, 32);
  v += __shfl_xor(v, 4, 32);
  v += __shfl_xor(v, 8, 32);
  return v;
}

#if HAVE_TDM
// TDM: DMA a [256 rows x 64 cols] f16 tile of V ([C][Nv] row-major) into LDS,
// packed contiguously as [256][64]. D# fields per CDNA5 ISA §8 (2-D descriptor,
// groups 2/3 and the trailing group zero => higher dims unused).
DEVINL void tdm_load_v_panel(const _Float16* gsrc, unsigned lds_addr) {
  unsigned long long ga = (unsigned long long)(size_t)gsrc;
  u32x4 g0 = {};
  g0[0] = 1u;                                    // count=1 (valid user descriptor)
  g0[1] = lds_addr;                              // lds_addr (bytes)
  g0[2] = (unsigned)ga;                          // global_addr[31:0]
  g0[3] = (unsigned)((ga >> 32) & 0x01FFFFFFu)   // global_addr[56:32]
        | (2u << 30);                            // type = 2 ("image")
  i32x8 g1 = {};
  g1[0] = (int)(1u << 16);        // workgroup_mask=0, data_size=1 (2 bytes)
  g1[1] = (int)(4096u << 16);     // tensor_dim0[15:0] = 4096 (elements)
  g1[2] = (int)(256u << 16);      // tensor_dim0[31:16]=0 | tensor_dim1[15:0]=256
  g1[3] = (int)(64u << 16);       // tensor_dim1[31:16]=0 | tile_dim0 = 64
  g1[4] = 256;                    // tile_dim1 = 256, tile_dim2 = 0
  g1[5] = 4096;                   // tensor_dim0_stride[31:0] = 4096 (elements)
  g1[6] = 0;                      // stride0[47:32]=0 | stride1[15:0]=0
  g1[7] = 0;
  i32x4 gz4 = {};
  i32x8 gz8 = {};
  // 6-arg form (clang-23 / therock-10.0 headers): (g0, g1, g2, g3, g4, cpol)
  __builtin_amdgcn_tensor_load_to_lds(g0, g1, gz4, gz4, gz8, 0);
}
#endif

// ---------------- conversion / packing kernels ----------------

// w [O][I][3][3][3] fp32 -> wh [t][o][c] f16 (A operand: K=c contiguous)
__global__ void k_cvt_convw(const float* __restrict__ w, _Float16* __restrict__ wh) {
  int idx = blockIdx.x * 256 + threadIdx.x;
  if (idx >= 27 * C * C) return;
  int t = idx >> 16;
  int o = (idx >> 8) & 255;
  int c = idx & 255;
  wh[idx] = (_Float16)w[((size_t)(o * C + c)) * 27 + t];
}

__global__ void k_cvt_f16(const float* __restrict__ src, _Float16* __restrict__ dst, int n) {
  int i = blockIdx.x * 256 + threadIdx.x;
  if (i < n) dst[i] = (_Float16)src[i];
}

__global__ void k_zero16(_Float16* __restrict__ dst, int n) {
  int i = blockIdx.x * 256 + threadIdx.x;
  if (i < n) dst[i] = (_Float16)0.f;
}

// x [b][c][n] fp32 -> xpad [b][18][18][18][c] f16 (zero halo, channel-last)
__global__ void k_build_xpad(const float* __restrict__ x, _Float16* __restrict__ xpad) {
  int p = blockIdx.x, b = blockIdx.y, c = threadIdx.x;
  int pw = p / (P1 * P1), rem = p % (P1 * P1), ph = rem / P1, pd = rem % P1;
  _Float16 v = (_Float16)0.f;
  if (pw >= 1 && pw <= 16 && ph >= 1 && ph <= 16 && pd >= 1 && pd <= 16) {
    int n = ((pw - 1) * 16 + (ph - 1)) * 16 + (pd - 1);
    v = (_Float16)x[((size_t)(b * C + c)) * Nv + n];
  }
  xpad[((size_t)(b * PV) + p) * C + c] = v;
}

// ---------------- conv3d as implicit GEMM ----------------
__global__ void k_conv(const _Float16* __restrict__ xpad,  // [B][PV][C] f16, halo = 0
                       const _Float16* __restrict__ wh,    // [27][C][C] f16
                       const float* __restrict__ bias,
                       float* __restrict__ y) {            // [B][C][Nv] fp32
  const int lane = threadIdx.x & 31;
  const int b  = blockIdx.z;
  const int m0 = blockIdx.y * 16;
  const int n0 = blockIdx.x * 64;
  const int nloc = lane & 15;
  const int kbB = (lane < 16) ? 0 : 8;
  const int kbA = (lane < 16) ? 0 : 8;

  const _Float16* bp[4];
#pragma unroll
  for (int s = 0; s < 4; ++s) {
    int n = n0 + s * 16 + nloc;
    int w = n >> 8, h = (n >> 4) & 15, d = n & 15;
    bp[s] = xpad + ((size_t)(b * PV) +
                    (size_t)((w + 1) * (P1 * P1) + (h + 1) * P1 + (d + 1))) * C + kbB;
  }
  const _Float16* ap = wh + (size_t)(m0 + nloc) * C + kbA;

  v8f acc0 = {}, acc1 = {}, acc2 = {}, acc3 = {};

#pragma unroll
  for (int t = 0; t < 27; ++t) {
    const int dw = t / 9 - 1, dh = (t / 3) % 3 - 1, dd = t % 3 - 1;
    const int toff = (dw * (P1 * P1) + dh * P1 + dd) * C;  // constant per unrolled tap
#pragma unroll
    for (int ck = 0; ck < 8; ++ck) {
      const int c0 = ck * 32;
      v16h A = cat16(*(const v8h*)(ap + (size_t)t * C * C + c0),
                     *(const v8h*)(ap + (size_t)t * C * C + c0 + 16));
#define CONV_STEP(s, ACC)                                           \
      {                                                             \
        const _Float16* q = bp[s] + toff + c0;                      \
        v16h Bt = cat16(*(const v8h*)q, *(const v8h*)(q + 16));     \
        ACC = wmma_f16(A, Bt, ACC);                                 \
      }
      CONV_STEP(0, acc0)
      CONV_STEP(1, acc1)
      CONV_STEP(2, acc2)
      CONV_STEP(3, acc3)
#undef CONV_STEP
    }
  }
  const int mb = (lane < 16) ? 0 : 8;
  float* yb = y + (size_t)b * C * Nv;
#pragma unroll
  for (int r = 0; r < 8; ++r) {
    int m = m0 + mb + r;
    float bv = bias[m];
    yb[(size_t)m * Nv + (n0 +  0 + nloc)] = acc0[r] + bv;
    yb[(size_t)m * Nv + (n0 + 16 + nloc)] = acc1[r] + bv;
    yb[(size_t)m * Nv + (n0 + 32 + nloc)] = acc2[r] + bv;
    yb[(size_t)m * Nv + (n0 + 48 + nloc)] = acc3[r] + bv;
  }
}

// ---------------- instance norm (per b,c over Nv) ----------------
__global__ void k_inorm(const float* __restrict__ y,
                        const float* __restrict__ xres,
                        _Float16* __restrict__ outp,
                        float* __restrict__ resf,
                        int mode) {
  int c = blockIdx.x, b = blockIdx.y, tid = threadIdx.x;
  const float* yp = y + ((size_t)(b * C + c)) * Nv;
  float s = 0.f, sq = 0.f;
  float vals[16];
#pragma unroll
  for (int i = 0; i < 16; ++i) {
    float v = yp[tid + i * 256];
    vals[i] = v; s += v; sq += v * v;
  }
  __shared__ float sh1[256], sh2[256];
  sh1[tid] = s; sh2[tid] = sq; __syncthreads();
  for (int off = 128; off > 0; off >>= 1) {
    if (tid < off) { sh1[tid] += sh1[tid + off]; sh2[tid] += sh2[tid + off]; }
    __syncthreads();
  }
  float mean = sh1[0] * (1.f / Nv);
  float var  = sh2[0] * (1.f / Nv) - mean * mean;
  float inv  = rsqrtf(var + 1e-5f);
#pragma unroll
  for (int i = 0; i < 16; ++i) {
    int n = tid + i * 256;
    float v = (vals[i] - mean) * inv;
    if (mode == 0) {
      v = fmaxf(v, 0.f);
      int w = n >> 8, h = (n >> 4) & 15, d = n & 15;
      size_t pidx = (size_t)(b * PV) +
                    (size_t)((w + 1) * (P1 * P1) + (h + 1) * P1 + (d + 1));
      outp[pidx * C + c] = (_Float16)v;
    } else {
      float r = xres[((size_t)(b * C + c)) * Nv + n] + v;
      resf[((size_t)(b * C + c)) * Nv + n] = r;
      outp[((size_t)(b * Nv + n)) * C + c] = (_Float16)r;
    }
  }
}

// ---------------- 1x1x1 convs (Q/K/V) ----------------
__global__ void k_linear(const _Float16* __restrict__ wh,  // [M][C] f16
                         const float* __restrict__ bias,   // [M]
                         const _Float16* __restrict__ xcl, // [B][Nv][C] f16
                         _Float16* __restrict__ out,
                         int M, int chan_last) {
  int lane = threadIdx.x & 31;
  int b = blockIdx.z, m0 = blockIdx.y * 16, n0 = blockIdx.x * 16;
  const _Float16* xb = xcl + (size_t)b * Nv * C + (size_t)n0 * C;
  v8f acc = {};
#pragma unroll
  for (int ck = 0; ck < 8; ++ck) {
    int c0 = ck * 32;
    v16h A  = load_A_rm(wh + (size_t)m0 * C + c0, C);
    v16h Bt = load_B_kc(xb + c0, C);
    acc = wmma_f16(A, Bt, acc);
  }
  int mb = (lane < 16) ? 0 : 8, nloc = lane & 15;
#pragma unroll
  for (int r = 0; r < 8; ++r) {
    int m = m0 + mb + r, n = n0 + nloc;
    float v = acc[r] + bias[m];
    if (chan_last) out[((size_t)(b * Nv + n)) * M + m] = (_Float16)v;  // q/k: [b][n][32]
    else           out[((size_t)(b * M + m)) * Nv + n] = (_Float16)v;  // v:   [b][c][n]
  }
}

// ---------------- flash attention + final residual ----------------
// One wave per 16-row block; per 64-col step: 4 score WMMAs + 32 PV WMMAs.
// V panel (256x64 f16 = 32KB) is TDM-DMA'd to LDS, double-buffered.
__global__ void k_flash(const _Float16* __restrict__ q16,  // [B][Nv][32]
                        const _Float16* __restrict__ k16,  // [B][Nv][32]
                        const _Float16* __restrict__ v16m, // [B][C][Nv]
                        const float* __restrict__ res,     // [B][C][Nv]
                        const float* __restrict__ gammaP,
                        float* __restrict__ out) {         // [B][C][Nv]
  __shared__ _Float16 Plds[16][64];
#if HAVE_TDM
  __shared__ _Float16 Vlds[2][256][64];
#endif
  int lane = threadIdx.x & 31;
  int b = blockIdx.y, i0 = blockIdx.x * 16;
  const int row_of = (lane < 16) ? 0 : 8;
  const int col = lane & 15;
  const int kbA = (lane < 16) ? 0 : 8;

  v16h Aq = load_A_rm(q16 + ((size_t)b * Nv + i0) * C8, C8);
  const _Float16* kb_ = k16 + (size_t)b * Nv * C8;
  const _Float16* vb_ = v16m + (size_t)b * C * Nv;

  float m_[8], l_[8];
#pragma unroll
  for (int r = 0; r < 8; ++r) { m_[r] = -3.0e38f; l_[r] = 0.f; }
  v8f acc[16];
  v8f vzero = {};
#pragma unroll
  for (int t = 0; t < 16; ++t) acc[t] = vzero;

#if HAVE_TDM
  const unsigned vldsa0 = (unsigned)(size_t)&Vlds[0][0][0];
  const unsigned vldsa1 = (unsigned)(size_t)&Vlds[1][0][0];
  tdm_load_v_panel(vb_, vldsa0);     // prefetch panel for j0 = 0
#endif

  for (int j0 = 0; j0 < Nv; j0 += 64) {
#if HAVE_TDM
    const int cur = (j0 >> 6) & 1;
    if (j0 + 64 < Nv) {
      // buffer we are about to fill was read in the previous step; make sure those
      // ds_loads drained before the DMA overwrites it
      asm volatile("s_wait_dscnt 0x0" ::: "memory");
      tdm_load_v_panel(vb_ + (j0 + 64), cur ? vldsa0 : vldsa1);
      __builtin_amdgcn_s_wait_tensorcnt(1);   // current panel complete
    } else {
      __builtin_amdgcn_s_wait_tensorcnt(0);
    }
    __syncthreads();
#endif
    v16h B0 = load_B_kc(kb_ + (size_t)(j0 +  0) * C8, C8);
    v16h B1 = load_B_kc(kb_ + (size_t)(j0 + 16) * C8, C8);
    v16h B2 = load_B_kc(kb_ + (size_t)(j0 + 32) * C8, C8);
    v16h B3 = load_B_kc(kb_ + (size_t)(j0 + 48) * C8, C8);
    v8f S0 = wmma_f16(Aq, B0, vzero);
    v8f S1 = wmma_f16(Aq, B1, vzero);
    v8f S2 = wmma_f16(Aq, B2, vzero);
    v8f S3 = wmma_f16(Aq, B3, vzero);

    float ef[8], p0[8], p1[8], p2[8], p3[8];
#pragma unroll
    for (int r = 0; r < 8; ++r) {
      float mx = fmaxf(fmaxf(S0[r], S1[r]), fmaxf(S2[r], S3[r]));
      mx = half_max16(mx);
      float mn = fmaxf(m_[r], mx);
      ef[r] = __expf(m_[r] - mn);
      m_[r] = mn;
      p0[r] = __expf(S0[r] - mn);
      p1[r] = __expf(S1[r] - mn);
      p2[r] = __expf(S2[r] - mn);
      p3[r] = __expf(S3[r] - mn);
      float rs = half_sum16((p0[r] + p1[r]) + (p2[r] + p3[r]));
      l_[r] = l_[r] * ef[r] + rs;
    }
#pragma unroll
    for (int t = 0; t < 16; ++t)
#pragma unroll
      for (int r = 0; r < 8; ++r) acc[t][r] *= ef[r];

    __syncthreads();  // protect Plds reads from previous iteration
#pragma unroll
    for (int r = 0; r < 8; ++r) {
      Plds[row_of + r][col]      = (_Float16)p0[r];
      Plds[row_of + r][col + 16] = (_Float16)p1[r];
      Plds[row_of + r][col + 32] = (_Float16)p2[r];
      Plds[row_of + r][col + 48] = (_Float16)p3[r];
    }
    __syncthreads();
    v16h Ap0 = cat16(*(const v8h*)&Plds[col][kbA],      *(const v8h*)&Plds[col][kbA + 16]);
    v16h Ap1 = cat16(*(const v8h*)&Plds[col][kbA + 32], *(const v8h*)&Plds[col][kbA + 48]);

#if HAVE_TDM
    const _Float16* vt = &Vlds[cur][0][0];
#pragma unroll
    for (int ct = 0; ct < 16; ++ct) {
      v16h Bv0 = load_B_kc(vt + (size_t)(ct * 16) * 64,      64);
      v16h Bv1 = load_B_kc(vt + (size_t)(ct * 16) * 64 + 32, 64);
      acc[ct] = wmma_f16(Ap0, Bv0, acc[ct]);
      acc[ct] = wmma_f16(Ap1, Bv1, acc[ct]);
    }
#else
#pragma unroll
    for (int ct = 0; ct < 16; ++ct) {
      v16h Bv0 = load_B_kc(vb_ + (size_t)(ct * 16) * Nv + j0,      Nv);
      v16h Bv1 = load_B_kc(vb_ + (size_t)(ct * 16) * Nv + j0 + 32, Nv);
      acc[ct] = wmma_f16(Ap0, Bv0, acc[ct]);
      acc[ct] = wmma_f16(Ap1, Bv1, acc[ct]);
    }
#endif
  }

  float g = gammaP[0];
#pragma unroll
  for (int ct = 0; ct < 16; ++ct) {
#pragma unroll
    for (int r = 0; r < 8; ++r) {
      int cch = ct * 16 + col;
      int i = i0 + row_of + r;
      size_t idx = ((size_t)b * C + cch) * Nv + i;
      out[idx] = g * (acc[ct][r] / l_[r]) + res[idx];
    }
  }
}

// ---------------- host side ----------------
extern "C" void kernel_launch(void* const* d_in, const int* in_sizes, int n_in,
                              void* d_out, int out_size, void* d_ws, size_t ws_size,
                              hipStream_t stream) {
  (void)in_sizes; (void)n_in; (void)out_size; (void)ws_size;
  const float* x     = (const float*)d_in[0];
  const float* w1    = (const float*)d_in[1];
  const float* b1    = (const float*)d_in[2];
  const float* w2    = (const float*)d_in[3];
  const float* b2    = (const float*)d_in[4];
  const float* wq    = (const float*)d_in[5];
  const float* bq    = (const float*)d_in[6];
  const float* wk    = (const float*)d_in[7];
  const float* bk    = (const float*)d_in[8];
  const float* wv    = (const float*)d_in[9];
  const float* bv    = (const float*)d_in[10];
  const float* gamma = (const float*)d_in[11];
  float* out = (float*)d_out;

  char* ws = (char*)d_ws;
  size_t off = 0;
  auto alloc = [&](size_t bytes) -> void* {
    void* p = ws + off;
    off += (bytes + 255) & ~(size_t)255;
    return p;
  };
  _Float16* xpad  = (_Float16*)alloc((size_t)Bc * PV * C * 2);
  _Float16* h1pad = (_Float16*)alloc((size_t)Bc * PV * C * 2);
  _Float16* w1h   = (_Float16*)alloc((size_t)27 * C * C * 2);
  _Float16* w2h   = (_Float16*)alloc((size_t)27 * C * C * 2);
  float*    ybuf  = (float*)   alloc((size_t)Bc * C * Nv * 4);
  float*    resf  = (float*)   alloc((size_t)Bc * C * Nv * 4);
  _Float16* rescl = (_Float16*)alloc((size_t)Bc * Nv * C * 2);
  _Float16* wqh   = (_Float16*)alloc((size_t)C8 * C * 2);
  _Float16* wkh   = (_Float16*)alloc((size_t)C8 * C * 2);
  _Float16* wvh   = (_Float16*)alloc((size_t)C * C * 2);
  _Float16* q16   = (_Float16*)alloc((size_t)Bc * Nv * C8 * 2);
  _Float16* k16   = (_Float16*)alloc((size_t)Bc * Nv * C8 * 2);
  _Float16* v16b  = (_Float16*)alloc((size_t)Bc * C * Nv * 2);

  k_cvt_convw<<<(27 * C * C + 255) / 256, 256, 0, stream>>>(w1, w1h);
  k_cvt_convw<<<(27 * C * C + 255) / 256, 256, 0, stream>>>(w2, w2h);
  k_cvt_f16<<<(C8 * C + 255) / 256, 256, 0, stream>>>(wq, wqh, C8 * C);
  k_cvt_f16<<<(C8 * C + 255) / 256, 256, 0, stream>>>(wk, wkh, C8 * C);
  k_cvt_f16<<<(C * C + 255) / 256, 256, 0, stream>>>(wv, wvh, C * C);
  {
    dim3 g(PV, Bc);
    k_build_xpad<<<g, 256, 0, stream>>>(x, xpad);
  }
  {
    int n = Bc * PV * C;
    k_zero16<<<(n + 255) / 256, 256, 0, stream>>>(h1pad, n);  // halo of conv2 input
  }
  dim3 gconv(Nv / 64, C / 16, Bc);
  dim3 gnorm(C, Bc);
  k_conv<<<gconv, 32, 0, stream>>>(xpad, w1h, b1, ybuf);
  k_inorm<<<gnorm, 256, 0, stream>>>(ybuf, nullptr, h1pad, nullptr, 0);
  k_conv<<<gconv, 32, 0, stream>>>(h1pad, w2h, b2, ybuf);
  k_inorm<<<gnorm, 256, 0, stream>>>(ybuf, x, rescl, resf, 1);

  dim3 gqk(Nv / 16, C8 / 16, Bc);
  k_linear<<<gqk, 32, 0, stream>>>(wqh, bq, rescl, q16, C8, 1);
  k_linear<<<gqk, 32, 0, stream>>>(wkh, bk, rescl, k16, C8, 1);
  dim3 gv(Nv / 16, C / 16, Bc);
  k_linear<<<gv, 32, 0, stream>>>(wvh, bv, rescl, v16b, C, 0);

  dim3 gf(Nv / 16, Bc);
  k_flash<<<gf, 32, 0, stream>>>(q16, k16, v16b, resf, gamma, out);
}
